// Gctu_10746008174711
// MI455X (gfx1250) — matrix operand order repaired and checked by reading, not compiled, not run on hardware
//
#include <hip/hip_runtime.h>
#include <hip/hip_bf16.h>

// ---------------------------------------------------------------------------
// Problem constants (from the reference)
// ---------------------------------------------------------------------------
constexpr int Bn = 4;          // batch
constexpr int Nn = 2048;       // sequence
constexpr int En = 512;        // embed
constexpr int Hh = 8;          // heads
constexpr int Kk = 128;        // vander rank
constexpr int Dd = 192;        // per-head dim
constexpr int D1 = Hh * Dd;    // 1536
constexpr int M2 = 2 * Nn - 1; // 4095 (toeplitz kernel length)
constexpr int NT = 4096;       // padded row stride for t / padded vander rows
constexpr int BN = Bn * Nn;    // 8192 flattened rows

typedef __attribute__((ext_vector_type(16))) __bf16 v16bf;
typedef __attribute__((ext_vector_type(8)))  __bf16 v8bf;
typedef __attribute__((ext_vector_type(8)))  float  v8f;
typedef __attribute__((ext_vector_type(4)))  unsigned int u32x4;
typedef __attribute__((ext_vector_type(8)))  int        i32x8;
typedef __attribute__((ext_vector_type(4)))  int        i32x4;

#define WMMA_BF16(a, b, c) \
  __builtin_amdgcn_wmma_f32_16x16x32_bf16(false, (a), false, (b), (short)0, (c), false, false)

__device__ __forceinline__ __bf16 f2bf(float f) {
  unsigned u = __float_as_uint(f);
  u += 0x7FFFu + ((u >> 16) & 1u);               // round-to-nearest-even
  unsigned short s = (unsigned short)(u >> 16);
  return __builtin_bit_cast(__bf16, s);
}

__device__ __forceinline__ float silu(float z) {
  return z / (1.f + __expf(-z));
}

__device__ __forceinline__ v16bf cat8(v8bf lo, v8bf hi) {
  return __builtin_shufflevector(lo, hi, 0, 1, 2, 3, 4, 5, 6, 7,
                                 8, 9, 10, 11, 12, 13, 14, 15);
}

// Load an A-fragment (16x32 bf16) from a row-major bf16 matrix, 16B-aligned.
// Layout: lane r=lane&15 holds row r; K = idx + 8*(idx>=8) + 8*(lane>=16).
__device__ __forceinline__ v16bf load_a(const __bf16* __restrict__ base, int ld,
                                        int row, int k0, int lane) {
  const __bf16* p = base + (size_t)row * ld + k0 + ((lane >> 4) << 3);
  return cat8(*(const v8bf*)p, *(const v8bf*)(p + 16));
}

// Load a B-fragment (32x16 bf16) whose column data is a contiguous row of a
// row-major matrix (i.e. B[k][n] = M[n][k]).  K = idx + 8*(idx>=8) + 8*(lane>=16).
__device__ __forceinline__ v16bf load_bt(const __bf16* __restrict__ base, int ld,
                                         int col, int k0, int lane) {
  const __bf16* p = base + (size_t)col * ld + k0 + ((lane >> 4) << 3);
  return cat8(*(const v8bf*)p, *(const v8bf*)(p + 16));
}

// ---------------------------------------------------------------------------
// Tensor Data Mover: 1-D contiguous copy global -> LDS (nelem 2-byte elems).
// D# built per CDNA5 ISA.
// ---------------------------------------------------------------------------
__device__ __forceinline__ void tdm_load_1d(const void* gptr, void* lptr,
                                            unsigned nelem) {
  unsigned long long ga = (unsigned long long)(uintptr_t)gptr;
  u32x4 g0;
  g0[0] = 1u;                                            // count=1, user mode
  g0[1] = (unsigned)(uintptr_t)lptr;                     // LDS byte address
  g0[2] = (unsigned)ga;                                  // global addr [31:0]
  g0[3] = (unsigned)((ga >> 32) & 0x01FFFFFFu) | (2u << 30);  // [56:32]|type=2
  i32x8 g1;
  g1[0] = (int)(1u << 16);                               // data_size = 2 bytes
  g1[1] = (int)((nelem & 0xFFFFu) << 16);                // tensor_dim0 lo16
  g1[2] = (int)(((nelem >> 16) & 0xFFFFu) | (1u << 16)); // dim0 hi | dim1=1 lo
  g1[3] = (int)((nelem & 0xFFFFu) << 16);                // dim1 hi=0 | tile_dim0
  g1[4] = 0;                                             // tile_dim1/2 unused
  g1[5] = (int)nelem;                                    // dim0_stride lo32
  g1[6] = 0;
  g1[7] = 0;
  i32x4 z4;
  z4[0] = 0; z4[1] = 0; z4[2] = 0; z4[3] = 0;
#if __has_include(<hip/amd_detail/amd_gfx1250_TDM.h>)
  i32x8 z8;
  z8[0] = 0; z8[1] = 0; z8[2] = 0; z8[3] = 0;
  z8[4] = 0; z8[5] = 0; z8[6] = 0; z8[7] = 0;
  __builtin_amdgcn_tensor_load_to_lds(g0, g1, z4, z4, z8, 0);
#else
  __builtin_amdgcn_tensor_load_to_lds(g0, g1, z4, z4, 0);
#endif
}

// ---------------------------------------------------------------------------
// Kernel 1: fp32 -> bf16 operand preparation.
// ---------------------------------------------------------------------------
__global__ void k_convert(const float* __restrict__ x,  const float* __restrict__ wu,
                          const float* __restrict__ wv, const float* __restrict__ wo,
                          const float* __restrict__ vander, const float* __restrict__ coef,
                          __bf16* __restrict__ xb,  __bf16* __restrict__ wub,
                          __bf16* __restrict__ wvb, __bf16* __restrict__ wob,
                          __bf16* __restrict__ vanb, __bf16* __restrict__ coefT) {
  int stride = gridDim.x * blockDim.x;
  int tid = blockIdx.x * blockDim.x + threadIdx.x;
  for (int i = tid; i < BN * En; i += stride) xb[i] = f2bf(x[i]);
  for (int i = tid; i < D1 * En; i += stride) {
    wub[i] = f2bf(wu[i]);
    wvb[i] = f2bf(wv[i]);
    wob[i] = f2bf(wo[i]);
  }
  for (int i = tid; i < NT * Kk; i += stride) {
    int m = i >> 7;
    vanb[i] = (m < M2) ? f2bf(vander[i]) : f2bf(0.f);
  }
  for (int i = tid; i < D1 * Kk; i += stride) {
    int hd = i >> 7;           // h*Dd + d
    int k  = i & (Kk - 1);
    int h  = hd / Dd;
    int d  = hd % Dd;
    coefT[i] = f2bf(coef[((size_t)h * Kk + k) * Dd + d]);
  }
}

// ---------------------------------------------------------------------------
// Kernel 2: t[h,m,d] = sum_k vander[m,k]*coef[h,k,d], stored transposed as
// tconv[h*Dd+d][m] (row stride NT).  One wave per 16(m) x 16(d) tile.
// ---------------------------------------------------------------------------
__global__ void k_tproj(const __bf16* __restrict__ vanb, const __bf16* __restrict__ coefT,
                        __bf16* __restrict__ tconv) {
  int lane = threadIdx.x & 31;
  int mt = blockIdx.x;      // 0..255
  int dt = blockIdx.y;      // 0..11
  int h  = blockIdx.z;      // 0..7
  int mrow = mt * 16 + (lane & 15);        // < 4096 (padded)
  int dcol = dt * 16 + (lane & 15);        // < 192

  v8f acc;
#pragma unroll
  for (int i = 0; i < 8; i++) acc[i] = 0.f;

#pragma unroll
  for (int k0 = 0; k0 < Kk; k0 += 32) {
    v16bf a = load_a(vanb, Kk, mrow, k0, lane);
    v16bf b = load_bt(coefT, Kk, h * Dd + dcol, k0, lane);
    acc = WMMA_BF16(a, b, acc);
  }

  int mbase = mt * 16 + ((lane >> 4) << 3);
  v8bf st;
#pragma unroll
  for (int rr = 0; rr < 8; rr++) st[rr] = f2bf(acc[rr]);
  // unconditional: m=4095 slot is inside the padded row and never read
  *(v8bf*)(tconv + ((size_t)h * Dd + dcol) * NT + mbase) = st;
}

// ---------------------------------------------------------------------------
// Kernel 3: u/v projection GEMM + fused SiLU.
// mode 0 -> u  fp32, layout [d1][b*N];  mode 1 -> v bf16, layout [d1][b*N]
// ---------------------------------------------------------------------------
__global__ void __launch_bounds__(256)
k_uv(const __bf16* __restrict__ xb, const __bf16* __restrict__ wb,
     const float* __restrict__ bias, float* __restrict__ u_out,
     __bf16* __restrict__ v_out, int mode) {
  int lane = threadIdx.x & 31;
  int wid  = threadIdx.x >> 5;
  int id = blockIdx.x * 8 + wid;
  int mt = id / (D1 / 16);   // 0..511
  int nt = id % (D1 / 16);   // 0..95
  int arow = mt * 16 + (lane & 15);
  int brow = nt * 16 + (lane & 15);

  v8f acc;
#pragma unroll
  for (int i = 0; i < 8; i++) acc[i] = 0.f;

#pragma unroll 4
  for (int k0 = 0; k0 < En; k0 += 32) {
    v16bf a = load_a(xb, En, arow, k0, lane);
    v16bf b = load_bt(wb, En, brow, k0, lane);
    acc = WMMA_BF16(a, b, acc);
  }

  int col   = nt * 16 + (lane & 15);
  float bia = bias[col];
  int rbase = mt * 16 + ((lane >> 4) << 3);
  if (mode == 0) {
    v8f res;
#pragma unroll
    for (int rr = 0; rr < 8; rr++) res[rr] = silu(acc[rr] + bia);
    *(v8f*)(u_out + (size_t)col * BN + rbase) = res;
  } else {
    v8bf res;
#pragma unroll
    for (int rr = 0; rr < 8; rr++) res[rr] = f2bf(silu(acc[rr] + bia));
    *(v8bf*)(v_out + (size_t)col * BN + rbase) = res;
  }
}

// ---------------------------------------------------------------------------
// Kernel 4: depthwise Toeplitz long-conv + fused u-gate.
//   out[b,h,i,d] = sum_j t[h, i-j+N-1, d] * v[b,h,j,d]
// One workgroup per channel d1.  TDM stages the t-row (8 KB) and v-slab
// (16 KB) into LDS once.  Each wave processes 8 output row-tiles at a time
// with an 8-slot rolling A-fragment file: fragment(q, J+1) == fragment(q-2, J)
// so each J-step loads 2 new A-fragments + 1 B-fragment for 8 WMMAs.  Slot
// rotation slot(q,J) = (q - 2J) mod 8 has period 4 -> J-loop unrolled in
// blocks of 4 phases so every register index is a compile-time constant.
// All LDS accesses index the __shared__ arrays directly (addrspace(3) ds ops).
// ---------------------------------------------------------------------------
__global__ void __launch_bounds__(256)
k_conv(const __bf16* __restrict__ tconv, const __bf16* __restrict__ vb,
       const float* __restrict__ u, __bf16* __restrict__ gb) {
  __shared__ __align__(16) __bf16 s_t[NT];       // 8 KB
  __shared__ __align__(16) __bf16 s_v[Bn * Nn];  // 16 KB
  int tid = threadIdx.x;
  int hd = blockIdx.x;  // == d1 = h*Dd + d

  if (tid < 32) {   // wave 0 drives the Tensor Data Mover
    tdm_load_1d(tconv + (size_t)hd * NT, s_t, NT);
    tdm_load_1d(vb + (size_t)hd * BN, s_v, BN);
    __builtin_amdgcn_s_wait_tensorcnt(0);
  }
  __syncthreads();

  int lane = tid & 31, wid = tid >> 5;
  int r = lane & 15;
  int kh8  = (lane >> 4) << 3;
  int kh16 = (lane >> 4) << 4;
  int bcol = lane & 15;
  int vofs = (bcol & (Bn - 1)) * Nn + kh8;       // base offset into s_v

  // Toeplitz A-fragment gather, direct on the shared array (keeps DS ops DS).
  auto load_toep = [&](int base) -> v16bf {
    v16bf a;
#pragma unroll
    for (int i = 0; i < 8; i++) a[i]     = s_t[base - i];
#pragma unroll
    for (int i = 0; i < 8; i++) a[8 + i] = s_t[base - 16 - i];
    return a;
  };
  // B-fragment: two 8-element bf16 runs (16 apart) from the v slab in LDS.
  auto load_bv = [&](int j0) -> v16bf {
    v16bf b;
#pragma unroll
    for (int i = 0; i < 16; i++) b[i] = s_v[vofs + j0 + i + (i & 8)];
    return b;
  };

  for (int g = 0; g < 2; g++) {
    int It0 = wid * 16 + g * 8;                   // 8 consecutive row-tiles
    int lofs = 16 * It0 + (Nn - 1) + r - kh8;     // per-lane base, q=J=0

    v8f acc[8];
#pragma unroll
    for (int q = 0; q < 8; q++)
#pragma unroll
      for (int i = 0; i < 8; i++) acc[q][i] = 0.f;

    v16bf A[8];
    // prologue: slots 2..7 hold fragments q=2..7 at J=0
#pragma unroll
    for (int q = 2; q < 8; q++) A[q] = load_toep(lofs + 16 * q);

    for (int Jb = 0; Jb < Nn / 32; Jb += 4) {
#pragma unroll
      for (int ph = 0; ph < 4; ph++) {
        int J = Jb + ph;
        int b0 = lofs - 32 * J;
        A[(0 - 2 * ph) & 7] = load_toep(b0);
        A[(1 - 2 * ph) & 7] = load_toep(b0 + 16);
        v16bf b = load_bv(J * 32);
#pragma unroll
        for (int q = 0; q < 8; q++)
          acc[q] = WMMA_BF16(A[(q - 2 * ph) & 7], b, acc[q]);
      }
    }

    if (bcol < Bn) {
#pragma unroll
      for (int q = 0; q < 8; q++) {
        int rbase = (It0 + q) * 16 + kh8;
        int n = bcol * Nn + rbase;                       // flattened (b, i)
        v8f ug = *(const v8f*)(u + (size_t)hd * BN + n); // contiguous gate read
#pragma unroll
        for (int rr = 0; rr < 8; rr++)
          gb[(size_t)(n + rr) * D1 + hd] = f2bf(ug[rr] * acc[q][rr]);
      }
    }
  }
}

// ---------------------------------------------------------------------------
// Kernel 5: output projection  out[n,e] = g[n,:] . Wo[e,:] + bo[e]
// ---------------------------------------------------------------------------
__global__ void __launch_bounds__(256)
k_out(const __bf16* __restrict__ gb, const __bf16* __restrict__ wob,
      const float* __restrict__ bo, float* __restrict__ out) {
  int lane = threadIdx.x & 31;
  int wid  = threadIdx.x >> 5;
  int id = blockIdx.x * 8 + wid;
  int mt = id / (En / 16);   // 0..511
  int et = id % (En / 16);   // 0..31
  int arow = mt * 16 + (lane & 15);
  int brow = et * 16 + (lane & 15);

  v8f acc;
#pragma unroll
  for (int i = 0; i < 8; i++) acc[i] = 0.f;

#pragma unroll 4
  for (int k0 = 0; k0 < D1; k0 += 32) {
    v16bf a = load_a(gb, D1, arow, k0, lane);
    v16bf b = load_bt(wob, D1, brow, k0, lane);
    acc = WMMA_BF16(a, b, acc);
  }

  int ecol = et * 16 + (lane & 15);
  float bia = bo[ecol];
  int rbase = mt * 16 + ((lane >> 4) << 3);
#pragma unroll
  for (int rr = 0; rr < 8; rr++) {
    out[(size_t)(rbase + rr) * En + ecol] = acc[rr] + bia;
  }
}

// ---------------------------------------------------------------------------
// Host-side launcher
// ---------------------------------------------------------------------------
extern "C" void kernel_launch(void* const* d_in, const int* in_sizes, int n_in,
                              void* d_out, int out_size, void* d_ws, size_t ws_size,
                              hipStream_t stream) {
  const float* x      = (const float*)d_in[0];
  const float* vander = (const float*)d_in[1];
  /* d_in[2] = index matrix: unused (Toeplitz indices computed arithmetically) */
  const float* coef   = (const float*)d_in[3];
  const float* Wu     = (const float*)d_in[4];
  const float* bu     = (const float*)d_in[5];
  const float* Wv     = (const float*)d_in[6];
  const float* bv     = (const float*)d_in[7];
  const float* Wo     = (const float*)d_in[8];
  const float* bo     = (const float*)d_in[9];
  float* out = (float*)d_out;

  char* ws = (char*)d_ws;
  size_t off = 0;
  auto alloc = [&](size_t bytes) -> char* {
    char* p = ws + off;
    off = (off + bytes + 255) & ~(size_t)255;
    return p;
  };
  __bf16* xb    = (__bf16*)alloc((size_t)BN * En * 2);     //  8.0 MB
  __bf16* wub   = (__bf16*)alloc((size_t)D1 * En * 2);     //  1.5 MB
  __bf16* wvb   = (__bf16*)alloc((size_t)D1 * En * 2);     //  1.5 MB
  __bf16* wob   = (__bf16*)alloc((size_t)En * D1 * 2);     //  1.5 MB
  __bf16* vanb  = (__bf16*)alloc((size_t)NT * Kk * 2);     //  1.0 MB
  __bf16* coefT = (__bf16*)alloc((size_t)D1 * Kk * 2);     //  0.4 MB
  __bf16* tconv = (__bf16*)alloc((size_t)D1 * NT * 2);     // 12.6 MB
  float*  ubuf  = (float*)alloc((size_t)D1 * BN * 4);      // 50.3 MB  [d1][n]
  __bf16* vbuf  = (__bf16*)alloc((size_t)D1 * BN * 2);     // 25.2 MB  [d1][n]
  __bf16* gbuf  = (__bf16*)alloc((size_t)BN * D1 * 2);     // 25.2 MB  [n][d1]

  // 1) fp32 -> bf16 operand prep (incl. padded vander, transposed coef)
  k_convert<<<1024, 256, 0, stream>>>(x, Wu, Wv, Wo, vander, coef,
                                      xb, wub, wvb, wob, vanb, coefT);

  // 2) t = vander @ coef (per head), stored transposed [h*Dd+d][m]
  k_tproj<<<dim3(256, 12, 8), 32, 0, stream>>>(vanb, coefT, tconv);

  // 3) u = silu(x Wu^T + bu) fp32 [d1][n] ; v = silu(x Wv^T + bv) bf16 [d1][n]
  k_uv<<<(BN / 16) * (D1 / 16) / 8, 256, 0, stream>>>(xb, wub, bu, ubuf, vbuf, 0);
  k_uv<<<(BN / 16) * (D1 / 16) / 8, 256, 0, stream>>>(xb, wvb, bv, ubuf, vbuf, 1);

  // 4) TDM-staged depthwise Toeplitz long-conv + fused u-gate -> g (bf16)
  k_conv<<<D1, 256, 0, stream>>>(tconv, vbuf, ubuf, gbuf);

  // 5) out = g @ Wo^T + bo
  k_out<<<(BN / 16) * (En / 16) / 8, 256, 0, stream>>>(gbuf, wob, bo, out);
}
